// NonLocalBlock_58437325029910
// MI455X (gfx1250) — matrix operand order repaired and checked by reading, not compiled, not run on hardware
//
#include <hip/hip_runtime.h>
#include <hip/hip_bf16.h>

// ---------------------------------------------------------------------------
// Non-local block (self-attention) for MI455X / gfx1250, wave32 + WMMA f16.
//   B=8, C=256, N=2048
// All GEMMs use v_wmma_f32_16x16x32_f16 (f16 inputs, f32 accum).
// 2-wide register blocking: every A (or B) tile is reused across two WMMAs,
// halving operand traffic (global + LDS) per matrix op.
// ---------------------------------------------------------------------------

typedef __attribute__((ext_vector_type(16))) _Float16 v16h;
typedef __attribute__((ext_vector_type(8)))  _Float16 v8h;
typedef __attribute__((ext_vector_type(8)))  float    v8f;

#define B_ 8
#define C_ 256
#define N_ 2048

// ---- WMMA operand loaders (wave32 layouts per CDNA5 ISA 7.12.2) -----------
// A-matrix 16x32 f16: lanes 0-15 -> M=lane, K base 0 ; lanes 16-31 -> M=lane-16,
// K base 8.  halves[0..7] = K kb+0..7, halves[8..15] = K kb+16..23.
// Source must be row-major [M][K] with leading dim `ld` (works for global & LDS).
__device__ __forceinline__ v16h load_a16(const _Float16* base, int ld, int row0, int k0) {
    int lane = (int)(threadIdx.x & 31u);
    int m  = lane & 15;
    int kb = (lane >> 4) * 8;
    const _Float16* p = base + (size_t)(row0 + m) * ld + k0 + kb;
    v8h lo = *(const v8h*)(p);        // K = kb .. kb+7
    v8h hi = *(const v8h*)(p + 16);   // K = kb+16 .. kb+23
    return __builtin_shufflevector(lo, hi, 0,1,2,3,4,5,6,7,8,9,10,11,12,13,14,15);
}

// B-matrix 32x16 f16: lane holds column col=lane&15, K base kb=(lane>>4)*8;
// halves[0..7] = K kb+0..7, halves[8..15] = K kb+16..23 (same as A).
// Source stored as [col][K] row-major (i.e. the
// transpose of the mathematical B), leading dim `ld`.
__device__ __forceinline__ v16h load_b16(const _Float16* base, int ld, int k0, int col0) {
    int lane = (int)(threadIdx.x & 31u);
    int c  = lane & 15;
    int kb = (lane >> 4) * 8;
    const _Float16* p = base + (size_t)(col0 + c) * ld + k0 + kb;
    v8h lo = *(const v8h*)(p);
    v8h hi = *(const v8h*)(p + 16);
    return __builtin_shufflevector(lo, hi, 0,1,2,3,4,5,6,7,8,9,10,11,12,13,14,15);
}

__device__ __forceinline__ v8f wmma_f16(v16h a, v16h b, v8f c) {
    return __builtin_amdgcn_wmma_f32_16x16x32_f16(false, a, false, b, (short)0, c, false, false);
}

// ---------------------------------------------------------------------------
// K0a: convert the four CxC weight matrices to f16 (layout unchanged [o][i]).
__global__ void k_convert_weights(const float* w0, const float* w1,
                                  const float* w2, const float* w3, _Float16* dst) {
    int i = blockIdx.x * blockDim.x + threadIdx.x;
    if (i < C_ * C_) {
        dst[0 * C_ * C_ + i] = (_Float16)w0[i];
        dst[1 * C_ * C_ + i] = (_Float16)w1[i];
        dst[2 * C_ * C_ + i] = (_Float16)w2[i];
        dst[3 * C_ * C_ + i] = (_Float16)w3[i];
    }
}

// K0b: xT[b][n][c] = (f16) x[b][c][n], 32x32 LDS tile transpose.
__global__ void k_transpose_x(const float* __restrict__ x, _Float16* __restrict__ xT) {
    __shared__ float t[32][33];
    int b  = blockIdx.z;
    int c0 = blockIdx.y * 32;
    int n0 = blockIdx.x * 32;
    int tx = threadIdx.x & 31;
    int ty = threadIdx.x >> 5;                  // 0..7
    const float* xb = x + (size_t)b * C_ * N_;
#pragma unroll
    for (int r = ty; r < 32; r += 8)
        t[r][tx] = xb[(size_t)(c0 + r) * N_ + n0 + tx];
    __syncthreads();
    _Float16* xTb = xT + (size_t)b * N_ * C_;
#pragma unroll
    for (int r = ty; r < 32; r += 8)
        xTb[(size_t)(n0 + r) * C_ + c0 + tx] = (_Float16)t[tx][r];
}

// ---------------------------------------------------------------------------
// K1: fused theta/phi/g projections.  One wave per 16x32 (or 32x16) tile pair.
//   theta/phi stored transposed  [B][N][C]  (for Q/K A- and B-operands)
//   g stored natural             [B][C][N]  (for P*V B-operand, K=m contiguous)
// grid.x = 3 jobs * 8 o-pairs;  A (or B) operand shared across the pair.
__global__ void k_proj_qkv(const _Float16* __restrict__ xT, const _Float16* __restrict__ wAll,
                           const float* __restrict__ b_th, const float* __restrict__ b_ph,
                           const float* __restrict__ b_g,
                           _Float16* __restrict__ thetaT, _Float16* __restrict__ phiT,
                           _Float16* __restrict__ gC) {
    int b   = blockIdx.z;
    int n0  = blockIdx.y * 16;
    int job = blockIdx.x >> 3;                  // 0=theta 1=phi 2=g
    int o0  = (blockIdx.x & 7) * 32;            // pair of 16-wide o-tiles
    int lane = (int)(threadIdx.x & 31u);
    int col = lane & 15, rb = (lane >> 4) * 8;
    const _Float16* xTb = xT + (size_t)b * N_ * C_;
    v8f acc0 = {}, acc1 = {};
    if (job < 2) {
        // outT[n][o] = sum_i xT[n][i] * w[o][i] + bias[o]; A shared across o-pair
        const _Float16* w = wAll + (size_t)job * C_ * C_;
#pragma unroll
        for (int k = 0; k < C_; k += 32) {
            v16h a  = load_a16(xTb, C_, n0, k);        // rows n, K=i (shared)
            v16h b0 = load_b16(w,   C_, k, o0);        // cols o0..o0+15
            v16h b1 = load_b16(w,   C_, k, o0 + 16);   // cols o0+16..o0+31
            acc0 = wmma_f16(a, b0, acc0);
            acc1 = wmma_f16(a, b1, acc1);
        }
        const float* bias = (job == 0 ? b_th : b_ph);
        float bv0 = bias[o0 + col], bv1 = bias[o0 + 16 + col];
        _Float16* dst = (job == 0 ? thetaT : phiT) + (size_t)b * N_ * C_;
#pragma unroll
        for (int v = 0; v < 8; v++) {
            dst[(size_t)(n0 + rb + v) * C_ + o0 + col]      = (_Float16)(acc0[v] + bv0);
            dst[(size_t)(n0 + rb + v) * C_ + o0 + 16 + col] = (_Float16)(acc1[v] + bv1);
        }
    } else {
        // g[o][n] = sum_i w_g[o][i] * xT[n][i] + b_g[o]; B shared across o-pair
        const _Float16* w = wAll + (size_t)2 * C_ * C_;
#pragma unroll
        for (int k = 0; k < C_; k += 32) {
            v16h a0 = load_a16(w,   C_, o0, k);        // rows o0..o0+15, K=i
            v16h a1 = load_a16(w,   C_, o0 + 16, k);   // rows o0+16..o0+31
            v16h bm = load_b16(xTb, C_, k, n0);        // cols n, K=i (shared)
            acc0 = wmma_f16(a0, bm, acc0);
            acc1 = wmma_f16(a1, bm, acc1);
        }
        _Float16* dst = gC + (size_t)b * C_ * N_;
#pragma unroll
        for (int v = 0; v < 8; v++) {
            dst[(size_t)(o0 + rb + v) * N_ + n0 + col]      = (_Float16)(acc0[v] + b_g[o0 + rb + v]);
            dst[(size_t)(o0 + 16 + rb + v) * N_ + n0 + col] = (_Float16)(acc1[v] + b_g[o0 + 16 + rb + v]);
        }
    }
}

// ---------------------------------------------------------------------------
// K2: attention for one 16-row n-tile per workgroup (8 waves, 256 threads).
// LDS: S f32[16][N] scores (128KB) + P f16[16][N] probs (64KB) + stats.
// Phase1: S = thetaT(n,:) . phiT(m,:)^T — wave computes a 16x32 m-strip,
//         reusing the theta A-tile across two phi B-tiles.
// Softmax: row max -> exp -> row sum (f32, numerically safe).
// Phase2: yT(n,c) = P . g^T — one P A-tile from LDS reused across two g
//         B-tiles (halves the dominant ds_load stream).
__global__ void k_attention(const _Float16* __restrict__ thetaT,
                            const _Float16* __restrict__ phiT,
                            const _Float16* __restrict__ gC,
                            _Float16* __restrict__ yT) {
    extern __shared__ char smem[];
    float*    S      = (float*)smem;                                   // 16*N f32
    _Float16* P      = (_Float16*)(smem + (size_t)16 * N_ * 4);        // 16*N f16
    float*    red    = (float*)(smem + (size_t)16 * N_ * 4 + (size_t)16 * N_ * 2); // 256
    float*    rowmax = red + 256;                                      // 16
    float*    rowsum = rowmax + 16;                                    // 16

    int b   = blockIdx.y;
    int n0  = blockIdx.x * 16;
    int tid = (int)threadIdx.x;
    int w = tid >> 5, lane = tid & 31;
    int col = lane & 15, rb = (lane >> 4) * 8;

    const _Float16* th = thetaT + (size_t)b * N_ * C_;
    const _Float16* ph = phiT   + (size_t)b * N_ * C_;

    // ---- Phase 1: raw scores into LDS (each wave: 8 m-pairs, 16 WMMA each)
    for (int p = w; p < N_ / 32; p += 8) {
        int m0 = p * 32;
        v8f acc0 = {}, acc1 = {};
#pragma unroll
        for (int k = 0; k < C_; k += 32) {
            v16h a  = load_a16(th, C_, n0, k);        // rows n, K=c (shared)
            v16h b0 = load_b16(ph, C_, k, m0);        // cols m0..m0+15
            v16h b1 = load_b16(ph, C_, k, m0 + 16);   // cols m0+16..m0+31
            acc0 = wmma_f16(a, b0, acc0);
            acc1 = wmma_f16(a, b1, acc1);
        }
#pragma unroll
        for (int v = 0; v < 8; v++) {
            S[(size_t)(rb + v) * N_ + m0 + col]      = acc0[v];
            S[(size_t)(rb + v) * N_ + m0 + 16 + col] = acc1[v];
        }
    }
    __syncthreads();

    // ---- row max (16 threads per row, then 16->1)
    {
        int r = tid >> 4, j0 = tid & 15;
        float m = -3.0e38f;
        for (int j = j0; j < N_; j += 16) m = fmaxf(m, S[(size_t)r * N_ + j]);
        red[tid] = m;
    }
    __syncthreads();
    if (tid < 16) {
        float m = red[tid * 16];
#pragma unroll
        for (int i = 1; i < 16; i++) m = fmaxf(m, red[tid * 16 + i]);
        rowmax[tid] = m;
        rowsum[tid] = 0.0f;
    }
    __syncthreads();

    // ---- exp + row sums, pack probabilities to f16
    for (int r = 0; r < 16; r++) {
        float mx = rowmax[r];
        float local = 0.0f;
        for (int j = tid; j < N_; j += 256) {
            float p = __expf(S[(size_t)r * N_ + j] - mx);
            P[(size_t)r * N_ + j] = (_Float16)p;
            local += p;
        }
        atomicAdd(&rowsum[r], local);   // ds_add_f32
    }
    __syncthreads();

    // ---- Phase 2: out = P * g^T  (each wave: one 16x32 c-strip, 128 WMMA)
    const _Float16* g = gC + (size_t)b * C_ * N_;
    _Float16* dst = yT + (size_t)b * N_ * C_;
    {
        int c0 = w * 32;                               // 8 waves x 32 c = 256
        v8f acc0 = {}, acc1 = {};
        for (int k = 0; k < N_; k += 32) {
            __builtin_prefetch(g + (size_t)(c0 + col) * N_ + k + 256, 0, 1);
            v16h a  = load_a16(P, N_, 0, k);           // rows n (LDS), K=m (shared)
            v16h b0 = load_b16(g, N_, k, c0);          // cols c0..c0+15
            v16h b1 = load_b16(g, N_, k, c0 + 16);     // cols c0+16..c0+31
            acc0 = wmma_f16(a, b0, acc0);
            acc1 = wmma_f16(a, b1, acc1);
        }
#pragma unroll
        for (int v = 0; v < 8; v++) {
            float inv = 1.0f / rowsum[rb + v];
            dst[(size_t)(n0 + rb + v) * C_ + c0 + col]      = (_Float16)(acc0[v] * inv);
            dst[(size_t)(n0 + rb + v) * C_ + c0 + 16 + col] = (_Float16)(acc1[v] * inv);
        }
    }
}

// ---------------------------------------------------------------------------
// K3: out[b][o][n] = sum_i wW[o][i]*y[b][n][i] + bW[o] + x[b][o][n]  (fp32 out)
// One wave per 32x16 (o-pair x n) tile; yT B-tile shared across the o-pair.
__global__ void k_final_proj(const _Float16* __restrict__ yT, const _Float16* __restrict__ wAll,
                             const float* __restrict__ bW, const float* __restrict__ x,
                             float* __restrict__ out) {
    int b  = blockIdx.z;
    int n0 = blockIdx.y * 16;
    int o0 = blockIdx.x * 32;                   // pair of 16-wide o-tiles
    int lane = (int)(threadIdx.x & 31u);
    int col = lane & 15, rb = (lane >> 4) * 8;
    const _Float16* w = wAll + (size_t)3 * C_ * C_;
    const _Float16* y = yT + (size_t)b * N_ * C_;
    v8f acc0 = {}, acc1 = {};
#pragma unroll
    for (int k = 0; k < C_; k += 32) {
        v16h a0 = load_a16(w, C_, o0, k);        // rows o0..o0+15, K=i
        v16h a1 = load_a16(w, C_, o0 + 16, k);   // rows o0+16..o0+31
        v16h bm = load_b16(y, C_, k, n0);        // cols n, K=i (shared)
        acc0 = wmma_f16(a0, bm, acc0);
        acc1 = wmma_f16(a1, bm, acc1);
    }
    const float* xb = x   + (size_t)b * C_ * N_;
    float*       ob = out + (size_t)b * C_ * N_;
#pragma unroll
    for (int v = 0; v < 8; v++) {
        int oA = o0 + rb + v;
        int oB = o0 + 16 + rb + v;
        ob[(size_t)oA * N_ + n0 + col] = acc0[v] + bW[oA] + xb[(size_t)oA * N_ + n0 + col];
        ob[(size_t)oB * N_ + n0 + col] = acc1[v] + bW[oB] + xb[(size_t)oB * N_ + n0 + col];
    }
}

// ---------------------------------------------------------------------------
extern "C" void kernel_launch(void* const* d_in, const int* in_sizes, int n_in,
                              void* d_out, int out_size, void* d_ws, size_t ws_size,
                              hipStream_t stream) {
    (void)in_sizes; (void)n_in; (void)out_size; (void)ws_size;
    const float* x       = (const float*)d_in[0];
    const float* w_theta = (const float*)d_in[1];
    const float* b_theta = (const float*)d_in[2];
    const float* w_phi   = (const float*)d_in[3];
    const float* b_phi   = (const float*)d_in[4];
    const float* w_g     = (const float*)d_in[5];
    const float* b_g     = (const float*)d_in[6];
    const float* w_W     = (const float*)d_in[7];
    const float* b_W     = (const float*)d_in[8];
    float* out = (float*)d_out;

    // workspace layout (f16 buffers, all generously aligned); total ~42.5 MB
    const size_t bnc = (size_t)B_ * N_ * C_ * sizeof(_Float16);   // 8 MiB each
    char* ws = (char*)d_ws;
    _Float16* xT     = (_Float16*)(ws + 0 * bnc);
    _Float16* thetaT = (_Float16*)(ws + 1 * bnc);
    _Float16* phiT   = (_Float16*)(ws + 2 * bnc);
    _Float16* gC     = (_Float16*)(ws + 3 * bnc);
    _Float16* yT     = (_Float16*)(ws + 4 * bnc);
    _Float16* wAll   = (_Float16*)(ws + 5 * bnc);                 // 4*C*C f16

    k_convert_weights<<<(C_ * C_ + 255) / 256, 256, 0, stream>>>(
        w_theta, w_phi, w_g, w_W, wAll);

    k_transpose_x<<<dim3(N_ / 32, C_ / 32, B_), 256, 0, stream>>>(x, xT);

    k_proj_qkv<<<dim3(24, N_ / 16, B_), 32, 0, stream>>>(
        xT, wAll, b_theta, b_phi, b_g, thetaT, phiT, gC);

    // dynamic LDS: 16*N f32 scores + 16*N f16 probs + reduction scratch
    // = 131072 + 65536 + 1152 = 197760 B  (fits the 320 KB WGP LDS on gfx1250)
    size_t smem = (size_t)16 * N_ * 4 + (size_t)16 * N_ * 2 + (256 + 32) * sizeof(float);
    k_attention<<<dim3(N_ / 16, B_), 256, smem, stream>>>(thetaT, phiT, gC, yT);

    k_final_proj<<<dim3(C_ / 32, N_ / 16, B_), 32, 0, stream>>>(yT, wAll, b_W, x, out);
}
